// MultiHeadAttention_4492535792584
// MI455X (gfx1250) — matrix operand order repaired and checked by reading, not compiled, not run on hardware
//
#include <hip/hip_runtime.h>

// ---------------------------------------------------------------------------
// MHA for MI455X (gfx1250).
//  - every matmul via v_wmma_f32_16x16x32_bf16 (fp32 accumulate)
//  - flash-attention online softmax (no S x S materialization)
//  - tile movement via global_load_async_to_lds_b128 + s_wait_asynccnt,
//    double-buffered in the GEMMs
//  - all fragment traffic as b128 loads (ISA 16-bit A/B layouts are
//    contiguous 16B runs per lane)
// ---------------------------------------------------------------------------

typedef __attribute__((ext_vector_type(16))) __bf16         bf16x16;
typedef __attribute__((ext_vector_type(8)))  float          f32x8;
typedef __attribute__((ext_vector_type(4)))  float          f32x4;
typedef __attribute__((ext_vector_type(4)))  unsigned int   u32x4;
typedef __attribute__((ext_vector_type(8)))  unsigned short u16x8;

constexpr int D_MODEL   = 1024;
constexpr int NUM_HEADS = 16;
constexpr int DK        = 64;
constexpr int B_        = 2;
constexpr int S_        = 2048;

__device__ inline unsigned short f32_to_bf16(float f) {
  return __builtin_bit_cast(unsigned short, (__bf16)f);   // RTNE via v_cvt
}

struct FragB16 { union { bf16x16 v; u32x4 q[2]; unsigned int u[8]; }; };

__device__ inline f32x8 wmma_bf16(const FragB16& a, const FragB16& b, f32x8 c) {
  return __builtin_amdgcn_wmma_f32_16x16x32_bf16(false, a.v, false, b.v,
                                                 (short)0, c, false, false);
}

__device__ inline f32x8 zero8() {
  f32x8 z = {0.f, 0.f, 0.f, 0.f, 0.f, 0.f, 0.f, 0.f};
  return z;
}

// A fragment (16x32 bf16): per lane two contiguous 16B runs.
// lanes 0-15: K {0..7,16..23}; lanes 16-31: K {8..15,24..31}.
__device__ inline void load_a_frag(FragB16& f, const unsigned short* rowbase, int half) {
  f.q[0] = *(const u32x4*)(rowbase + half * 8);
  f.q[1] = *(const u32x4*)(rowbase + half * 8 + 16);
}
// B fragment (32x16 bf16) from [n][k] LDS: lane%16 = N, K {0..7,16..23} (half 0)
// or {8..15,24..31} (half 1) -> two contiguous 16B runs per lane.
__device__ inline void load_b_frag(FragB16& f, const unsigned short* colbase, int half) {
  f.q[0] = *(const u32x4*)(colbase + half * 8);
  f.q[1] = *(const u32x4*)(colbase + half * 8 + 16);
}

// --- async copy: global -> LDS, 16B per lane, ASYNCcnt tracked -------------
__device__ inline unsigned int lds_off(const void* p) {
  return (unsigned int)(unsigned long long)p;   // low 32 bits of flat LDS addr
}
__device__ inline void async_b128(const unsigned short* g, unsigned int lds) {
  asm volatile("global_load_async_to_lds_b128 %0, %1, off"
               :: "v"(lds), "v"((unsigned long long)g) : "memory");
}
__device__ inline void wait_async(int pending) {
  if (pending) asm volatile("s_wait_asynccnt 0x2" ::: "memory");
  else         asm volatile("s_wait_asynccnt 0x0" ::: "memory");
}

// 16-lane (half-wave) reductions via ds_swizzle xor 1,2,4,8.
#define SWZ1(x) __builtin_amdgcn_ds_swizzle((x), (0x1f | (1 << 10)))
#define SWZ2(x) __builtin_amdgcn_ds_swizzle((x), (0x1f | (2 << 10)))
#define SWZ4(x) __builtin_amdgcn_ds_swizzle((x), (0x1f | (4 << 10)))
#define SWZ8(x) __builtin_amdgcn_ds_swizzle((x), (0x1f | (8 << 10)))

__device__ inline float red16_max(float x) {
  x = fmaxf(x, __builtin_bit_cast(float, SWZ1(__builtin_bit_cast(int, x))));
  x = fmaxf(x, __builtin_bit_cast(float, SWZ2(__builtin_bit_cast(int, x))));
  x = fmaxf(x, __builtin_bit_cast(float, SWZ4(__builtin_bit_cast(int, x))));
  x = fmaxf(x, __builtin_bit_cast(float, SWZ8(__builtin_bit_cast(int, x))));
  return x;
}
__device__ inline float red16_sum(float x) {
  x += __builtin_bit_cast(float, SWZ1(__builtin_bit_cast(int, x)));
  x += __builtin_bit_cast(float, SWZ2(__builtin_bit_cast(int, x)));
  x += __builtin_bit_cast(float, SWZ4(__builtin_bit_cast(int, x)));
  x += __builtin_bit_cast(float, SWZ8(__builtin_bit_cast(int, x)));
  return x;
}

// ---------------------------------------------------------------------------
// Kernel 1: flat fp32 -> bf16 conversion (8 elements / thread, b128 traffic).
// ---------------------------------------------------------------------------
__global__ __launch_bounds__(256) void convert_bf16(
    const float* __restrict__ src, unsigned short* __restrict__ dst, int n8) {
  int i = blockIdx.x * 256 + threadIdx.x;
  if (i >= n8) return;
  const f32x4* s = (const f32x4*)(src + (size_t)i * 8);
  f32x4 a = s[0], b = s[1];
  u16x8 o;
  o[0] = f32_to_bf16(a.x); o[1] = f32_to_bf16(a.y);
  o[2] = f32_to_bf16(a.z); o[3] = f32_to_bf16(a.w);
  o[4] = f32_to_bf16(b.x); o[5] = f32_to_bf16(b.y);
  o[6] = f32_to_bf16(b.z); o[7] = f32_to_bf16(b.w);
  *(u16x8*)(dst + (size_t)i * 8) = o;
}

// ---------------------------------------------------------------------------
// Kernels 2/4: Y = X(bf16) @ W^T(bf16) + bias.  W kept [n][k] (its natural
// layout IS the B-fragment layout), so A and B tiles are straight async
// copies.  Block 64x64, BK=32, 8 waves, double-buffered LDS.
// ---------------------------------------------------------------------------
template <bool TO_HEADS>
__device__ inline void gemm_body(const unsigned short* __restrict__ X,
                                 const unsigned short* __restrict__ W,
                                 const float* __restrict__ bias,
                                 unsigned short* out_bf16, float* out_f32) {
  __shared__ __align__(16) unsigned short Alds[2][64][48];
  __shared__ __align__(16) unsigned short Blds[2][64][48];
  const int tid  = threadIdx.x;
  const int wave = tid >> 5, lane = tid & 31, half = lane >> 4, l16 = lane & 15;
  const int mr = wave & 3, cg = (wave >> 2) * 32;
  const int m0 = blockIdx.x * 64, n0 = blockIdx.y * 64;
  const int r = tid >> 2, seg = (tid & 3) * 8;          // 16B segment per thread
  f32x8 acc0 = zero8(), acc1 = zero8();

  const unsigned short* ga = X + (size_t)(m0 + r) * D_MODEL + seg;
  const unsigned short* gb = W + (size_t)(n0 + r) * D_MODEL + seg;

  // prologue: tile 0 into buffer 0
  async_b128(ga, lds_off(&Alds[0][r][seg]));
  async_b128(gb, lds_off(&Blds[0][r][seg]));

  for (int kt = 0; kt < D_MODEL / 32; ++kt) {
    const int buf = kt & 1;
    const bool more = (kt + 1) < (D_MODEL / 32);
    if (more) {
      async_b128(ga + (kt + 1) * 32, lds_off(&Alds[buf ^ 1][r][seg]));
      async_b128(gb + (kt + 1) * 32, lds_off(&Blds[buf ^ 1][r][seg]));
    }
    wait_async(more ? 1 : 0);
    __syncthreads();                       // tile kt visible to all waves

    FragB16 a, b0, b1;
    load_a_frag(a, &Alds[buf][mr * 16 + l16][0], half);
    load_b_frag(b0, &Blds[buf][cg + l16][0], half);
    load_b_frag(b1, &Blds[buf][cg + 16 + l16][0], half);
    acc0 = wmma_bf16(a, b0, acc0);
    acc1 = wmma_bf16(a, b1, acc1);
    __syncthreads();                       // done reading buf before refill
  }

  const int nc0 = n0 + cg + l16, nc1 = nc0 + 16;
  const float bv0 = bias[nc0], bv1 = bias[nc1];
#pragma unroll
  for (int j = 0; j < 8; ++j) {
    int m = m0 + mr * 16 + j + 8 * half;
    if (TO_HEADS) {                        // bf16 out in [B,H,S,dk]
      int bb = m >> 11, s = m & 2047;
      int h0 = nc0 >> 6, d0 = nc0 & 63, h1 = nc1 >> 6, d1 = nc1 & 63;
      out_bf16[(((size_t)bb * NUM_HEADS + h0) * S_ + s) * DK + d0] = f32_to_bf16(acc0[j] + bv0);
      out_bf16[(((size_t)bb * NUM_HEADS + h1) * S_ + s) * DK + d1] = f32_to_bf16(acc1[j] + bv1);
    } else {                               // fp32 out, row-major
      out_f32[(size_t)m * D_MODEL + nc0] = acc0[j] + bv0;
      out_f32[(size_t)m * D_MODEL + nc1] = acc1[j] + bv1;
    }
  }
}

__global__ __launch_bounds__(256) void gemm_qkv(
    const unsigned short* __restrict__ X, const unsigned short* __restrict__ W,
    const float* __restrict__ bias, unsigned short* __restrict__ out) {
  gemm_body<true>(X, W, bias, out, nullptr);
}
__global__ __launch_bounds__(256) void gemm_out(
    const unsigned short* __restrict__ X, const unsigned short* __restrict__ W,
    const float* __restrict__ bias, float* __restrict__ out) {
  gemm_body<false>(X, W, bias, nullptr, out);
}

// ---------------------------------------------------------------------------
// Kernel 3: flash attention per (b,h). 4 waves x 16 query rows; 64-key tiles.
// K tile is a straight copy -> async b128; V needs transpose -> DS stores.
// ---------------------------------------------------------------------------
__global__ __launch_bounds__(128) void attention(
    const unsigned short* __restrict__ Q, const unsigned short* __restrict__ K,
    const unsigned short* __restrict__ V, unsigned short* __restrict__ Cmb) {
  __shared__ __align__(16) unsigned short Klds[64][72];   // [key][d]
  __shared__ __align__(16) unsigned short VldsT[64][72];  // [d][key]
  __shared__ __align__(16) unsigned short Plds[4][16][64];

  const int tid  = threadIdx.x;
  const int wave = tid >> 5, lane = tid & 31, half = lane >> 4, l16 = lane & 15;
  const int bh = blockIdx.y;
  const int b  = bh >> 4, h = bh & 15;
  const int q0 = blockIdx.x * 64;
  const int qrow = q0 + wave * 16 + l16;

  // Q fragments (16x64 per wave = two 16x32 A frags) via global b128 loads.
  FragB16 qf[2];
  const unsigned short* qptr = Q + ((size_t)bh * S_ + qrow) * DK;
#pragma unroll
  for (int f = 0; f < 2; ++f) {
    qf[f].q[0] = *(const u32x4*)(qptr + f * 32 + half * 8);
    qf[f].q[1] = *(const u32x4*)(qptr + f * 32 + half * 8 + 16);
  }

  float mprev[8], lsum[8];
  f32x8 oacc[4];
#pragma unroll
  for (int j = 0; j < 8; ++j) { mprev[j] = -3.0e38f; lsum[j] = 0.f; }
#pragma unroll
  for (int t = 0; t < 4; ++t) oacc[t] = zero8();

  const unsigned short* kg = K + (size_t)bh * S_ * DK;
  const unsigned short* vg = V + (size_t)bh * S_ * DK;

  for (int kb = 0; kb < S_ / 64; ++kb) {
    __syncthreads();
    // K tile: async global->LDS, 512 x 16B segments
#pragma unroll
    for (int t = 0; t < 4; ++t) {
      int idx = tid + 128 * t;
      int kk = idx >> 3, seg = (idx & 7) * 8;
      async_b128(kg + (size_t)(kb * 64 + kk) * DK + seg, lds_off(&Klds[kk][seg]));
    }
    // V tile: transpose into [d][key]
#pragma unroll
    for (int i = tid; i < 2048; i += 128) {
      int kk = i >> 5, dp = (i & 31) * 2;
      unsigned int uv = *(const unsigned int*)(vg + (size_t)(kb * 64 + kk) * DK + dp);
      VldsT[dp][kk]     = (unsigned short)(uv & 0xffffu);
      VldsT[dp + 1][kk] = (unsigned short)(uv >> 16);
    }
    wait_async(0);
    __syncthreads();

    // scores: 16 rows x 64 keys, scaled by 1/sqrt(dk)
    f32x8 sc[4];
#pragma unroll
    for (int ct = 0; ct < 4; ++ct) {
      FragB16 kf0, kf1;
      const unsigned short* kbse = &Klds[ct * 16 + l16][0];
      load_b_frag(kf0, kbse, half);        // d 0..31
      load_b_frag(kf1, kbse + 32, half);   // d 32..63
      f32x8 a = zero8();
      a = wmma_bf16(qf[0], kf0, a);
      a = wmma_bf16(qf[1], kf1, a);
#pragma unroll
      for (int j = 0; j < 8; ++j) a[j] = a[j] * 0.125f;
      sc[ct] = a;
    }

    // online softmax (rows live in 16-lane half-wave groups)
#pragma unroll
    for (int j = 0; j < 8; ++j) {
      float mx = fmaxf(fmaxf(sc[0][j], sc[1][j]), fmaxf(sc[2][j], sc[3][j]));
      mx = red16_max(mx);
      float mnew  = fmaxf(mprev[j], mx);
      float alpha = __expf(mprev[j] - mnew);
      float rs = 0.f;
#pragma unroll
      for (int ct = 0; ct < 4; ++ct) {
        float p = __expf(sc[ct][j] - mnew);
        sc[ct][j] = p;
        rs += p;
      }
      rs = red16_sum(rs);
      lsum[j]  = lsum[j] * alpha + rs;
      mprev[j] = mnew;
#pragma unroll
      for (int ct = 0; ct < 4; ++ct) oacc[ct][j] = oacc[ct][j] * alpha;
    }

    // P: C layout -> per-wave LDS -> A-fragment layout
#pragma unroll
    for (int ct = 0; ct < 4; ++ct)
#pragma unroll
      for (int j = 0; j < 8; ++j)
        Plds[wave][j + 8 * half][ct * 16 + l16] = f32_to_bf16(sc[ct][j]);

    FragB16 pf0, pf1;
    const unsigned short* pb = &Plds[wave][l16][0];
    load_a_frag(pf0, pb, half);            // keys 0..31
    load_a_frag(pf1, pb + 32, half);       // keys 32..63
#pragma unroll
    for (int ct = 0; ct < 4; ++ct) {
      FragB16 vf0, vf1;
      const unsigned short* vb = &VldsT[ct * 16 + l16][0];
      load_b_frag(vf0, vb, half);
      load_b_frag(vf1, vb + 32, half);
      oacc[ct] = wmma_bf16(pf0, vf0, oacc[ct]);
      oacc[ct] = wmma_bf16(pf1, vf1, oacc[ct]);
    }
  }

  // normalize, write combined [B,S,D] bf16
#pragma unroll
  for (int j = 0; j < 8; ++j) {
    float inv = 1.0f / lsum[j];
    int s = q0 + wave * 16 + j + 8 * half;
#pragma unroll
    for (int ct = 0; ct < 4; ++ct) {
      int col = h * DK + ct * 16 + l16;
      Cmb[((size_t)b * S_ + s) * D_MODEL + col] = f32_to_bf16(oacc[ct][j] * inv);
    }
  }
}

// ---------------------------------------------------------------------------
extern "C" void kernel_launch(void* const* d_in, const int* in_sizes, int n_in,
                              void* d_out, int out_size, void* d_ws, size_t ws_size,
                              hipStream_t stream) {
  const float* query = (const float*)d_in[0];
  const float* key_  = (const float*)d_in[1];
  const float* value = (const float*)d_in[2];
  const float* W_q   = (const float*)d_in[3];
  const float* b_q   = (const float*)d_in[4];
  const float* W_k   = (const float*)d_in[5];
  const float* b_k   = (const float*)d_in[6];
  const float* W_v   = (const float*)d_in[7];
  const float* b_v   = (const float*)d_in[8];
  const float* W_o   = (const float*)d_in[9];
  const float* b_o   = (const float*)d_in[10];

  unsigned short* ws = (unsigned short*)d_ws;
  const size_t DD   = (size_t)D_MODEL * D_MODEL;   // 1M elems
  const size_t BHSD = (size_t)B_ * S_ * D_MODEL;   // 4M elems
  unsigned short* wq = ws;                   // bf16 W, natural [n][k] layout
  unsigned short* wk = ws + DD;
  unsigned short* wv = ws + 2 * DD;
  unsigned short* wo = ws + 3 * DD;
  unsigned short* xq = ws + 4 * DD;          // bf16 activations [B*S][D]
  unsigned short* xk = xq + BHSD;
  unsigned short* xv = xk + BHSD;
  unsigned short* qb = xv + BHSD;            // head-split [B,H,S,dk]
  unsigned short* kb = qb + BHSD;
  unsigned short* vb = kb + BHSD;
  unsigned short* cb = xq;                   // reuse: xq dead after gemm_qkv #1

  const int nW8 = (int)(DD / 8), nX8 = (int)(BHSD / 8);
  convert_bf16<<<dim3(nW8 / 256), 256, 0, stream>>>(W_q, wq, nW8);
  convert_bf16<<<dim3(nW8 / 256), 256, 0, stream>>>(W_k, wk, nW8);
  convert_bf16<<<dim3(nW8 / 256), 256, 0, stream>>>(W_v, wv, nW8);
  convert_bf16<<<dim3(nW8 / 256), 256, 0, stream>>>(W_o, wo, nW8);
  convert_bf16<<<dim3(nX8 / 256), 256, 0, stream>>>(query, xq, nX8);
  convert_bf16<<<dim3(nX8 / 256), 256, 0, stream>>>(key_,  xk, nX8);
  convert_bf16<<<dim3(nX8 / 256), 256, 0, stream>>>(value, xv, nX8);

  gemm_qkv<<<dim3(64, 16), 256, 0, stream>>>(xq, wq, b_q, qb);
  gemm_qkv<<<dim3(64, 16), 256, 0, stream>>>(xk, wk, b_k, kb);
  gemm_qkv<<<dim3(64, 16), 256, 0, stream>>>(xv, wv, b_v, vb);
  attention<<<dim3(S_ / 64, B_ * NUM_HEADS), 128, 0, stream>>>(qb, kb, vb, cb);
  gemm_out<<<dim3(64, 16), 256, 0, stream>>>(cb, wo, b_o, (float*)d_out);

  (void)in_sizes; (void)n_in; (void)out_size; (void)ws_size;
}
